// MyConv1d_ell_2_single_64037962383417
// MI455X (gfx1250) — matrix operand order repaired and checked by reading, not compiled, not run on hardware
//
#include <hip/hip_runtime.h>
#include <hip/hip_bf16.h>

// ---------------------------------------------------------------------------
// Problem constants (from the reference)
// ---------------------------------------------------------------------------
#define BATCH 256
#define CIN   1024
#define COUT  1024
#define LIN   18
#define NCOL  54                 // 9 t-slots * 6 lengths

typedef __attribute__((ext_vector_type(16))) __bf16 v16bf;
typedef __attribute__((ext_vector_type(8)))  __bf16 v8bf;
typedef __attribute__((ext_vector_type(8)))  float  v8f;

// Per-length K-unit tables: each unit is (x tap offset, weight plane).
// Planes 0..1 = filter_2 taps (scaled by 1/sqrt(2048)),
// planes 2..4 = filter_3 taps (scaled by 1/sqrt(3072)).
__constant__ int c_U[6] = {2, 4, 5, 9, 9, 3};
__constant__ int c_tap[6][9] = {
    {0, 4, 0, 0, 0,  0, 0, 0, 0},     // len4: (2,2)
    {0, 5, 1, 6, 0,  0, 0, 0, 0},     // len5: (2,3) (3,2)
    {1, 7, 0, 4, 8,  0, 0, 0, 0},     // len6: (3,3) (2,2,2)
    {0, 4, 9, 0, 5, 10, 1, 6, 10},    // len7: (2,2,3)(2,3,2)(3,2,2)
    {0, 5, 11, 1, 6, 11, 1, 7, 12},   // len8: (2,3,3)(3,2,3)(3,3,2)
    {1, 7, 13, 0, 0,  0, 0, 0, 0}     // len9: (3,3,3)
};
__constant__ int c_pl[6][9] = {
    {0, 1, 0, 0, 0, 0, 0, 0, 0},
    {0, 1, 0, 1, 0, 0, 0, 0, 0},
    {0, 1, 2, 3, 4, 0, 0, 0, 0},
    {2, 3, 4, 2, 3, 4, 2, 3, 4},
    {2, 3, 4, 2, 3, 4, 2, 3, 4},
    {2, 3, 4, 0, 0, 0, 0, 0, 0}
};

// ---------------------------------------------------------------------------
// CDNA5 async global->LDS copy (128b), with portable fallback.
// Probe-derived signature: param0 = non-const v4i32 in addrspace(1) (global),
// param1 = v4i32 in addrspace(3) (LDS), then imm offset, imm cpol.
// ---------------------------------------------------------------------------
typedef int async_v4i __attribute__((vector_size(16)));

#if __has_builtin(__builtin_amdgcn_global_load_async_to_lds_b128)
#define HAVE_ASYNC 1
#define ASYNC_COPY(dst, src)                                                 \
    __builtin_amdgcn_global_load_async_to_lds_b128(                          \
        (__attribute__((address_space(1))) async_v4i*)(uintptr_t)(src),      \
        (__attribute__((address_space(3))) async_v4i*)(dst), 0, 0)
#else
#define HAVE_ASYNC 0
#define ASYNC_COPY(dst, src) (*(uint4*)(dst) = *(const uint4*)(src))
#endif

__device__ __forceinline__ void wait_async_0() {
#if HAVE_ASYNC
#if __has_builtin(__builtin_amdgcn_s_wait_asynccnt)
    __builtin_amdgcn_s_wait_asynccnt(0);
#else
    asm volatile("s_wait_asynccnt 0x0" ::: "memory");
#endif
#endif
}

// ---------------------------------------------------------------------------
// Prep: x [B,C,L] f32  ->  xp [L][B][C] bf16 (hi + residual lo)
// ---------------------------------------------------------------------------
__global__ void split_x_kernel(const float* __restrict__ x,
                               __bf16* __restrict__ xh,
                               __bf16* __restrict__ xl, int n) {
    int i = blockIdx.x * blockDim.x + threadIdx.x;
    if (i >= n) return;
    int c = i & (CIN - 1);
    int b = (i >> 10) & (BATCH - 1);
    int l = i >> 18;                       // / (1024*256)
    float v = x[((size_t)b * CIN + c) * LIN + l];
    __bf16 h = (__bf16)v;
    xh[i] = h;
    xl[i] = (__bf16)(v - (float)h);
}

// ---------------------------------------------------------------------------
// Prep: filters -> wp [5 planes][O][C] bf16 hi/lo with scale folded in
// ---------------------------------------------------------------------------
__global__ void split_w_kernel(const float* __restrict__ f2,
                               const float* __restrict__ f3,
                               __bf16* __restrict__ wh,
                               __bf16* __restrict__ wl, int n) {
    int i = blockIdx.x * blockDim.x + threadIdx.x;
    if (i >= n) return;
    const float S2 = 0.022097086912079608f;   // 1/sqrt(2048)
    const float S3 = 0.018042195912175804f;   // 1/sqrt(3072)
    int c = i & (CIN - 1);
    int o = (i >> 10) & (COUT - 1);
    int p = i >> 20;
    float v;
    if (p < 2) v = f2[((size_t)o * CIN + c) * 2 + p] * S2;
    else       v = f3[((size_t)o * CIN + c) * 3 + (p - 2)] * S3;
    __bf16 h = (__bf16)v;
    wh[i] = h;
    wl[i] = (__bf16)(v - (float)h);
}

__global__ void zero_kernel(float4* __restrict__ out, int n4) {
    int i = blockIdx.x * blockDim.x + threadIdx.x;
    if (i < n4) out[i] = make_float4(0.f, 0.f, 0.f, 0.f);
}

// ---------------------------------------------------------------------------
// Fused WMMA GEMM over all 6 lengths.
// Block: 256 threads (8 waves). Tile: 128 (M=b) x 128 (N=o), one (li,t) each.
// Wave w: wm=w&3 -> 32-row M quarter, wn=w>>2 -> 64-col N half; 2x4 subtiles.
// bf16 split: acc += ah*bh + ah*bl + al*bh  (3 WMMAs/subtile/K-slice).
// Double-buffered LDS, async global->LDS copies overlapped with WMMA.
// ---------------------------------------------------------------------------
__global__ __launch_bounds__(256)
void gemm_wmma_kernel(const __bf16* __restrict__ xh, const __bf16* __restrict__ xl,
                      const __bf16* __restrict__ wh, const __bf16* __restrict__ wl,
                      float* __restrict__ out) {
    // 2 x (A 128x32 hi/lo + B 128x32 hi/lo) = 64 KB
    __shared__ alignas(16) __bf16 sAh[2][128 * 32];
    __shared__ alignas(16) __bf16 sAl[2][128 * 32];
    __shared__ alignas(16) __bf16 sBh[2][128 * 32];
    __shared__ alignas(16) __bf16 sBl[2][128 * 32];

    // decode blockIdx.x -> (li, t, m-half): lengths have 2*T m-tiles each
    int bx = blockIdx.x;
    int li = 0;
    for (;;) {
        const int tiles = 2 * (6 - li);
        if (bx < tiles) break;
        bx -= tiles;
        ++li;
    }
    const int t  = bx >> 1;
    const int b0 = (bx & 1) * 128;
    const int o0 = blockIdx.y * 128;

    const int tid  = threadIdx.x;
    const int lane = tid & 31;
    const int w    = tid >> 5;            // wave 0..7
    const int wm   = w & 3;               // 32-row M quarter
    const int wn   = w >> 2;              // 64-col N half
    const int hl   = lane >> 4;           // lane half (K-half selector)
    const int l16  = lane & 15;

    // cooperative-load mapping: 128 rows x 4 chunks of 8 bf16; 2 chunks/thread
    const int r2 = tid >> 1;
    const int k2 = (tid & 1) * 2;

    v8f acc[2][4] = {};

    const int U = c_U[li];
    const int S = U * 32;                 // K-slices of 32 channels

    auto issue = [&](int s, int buf) {
        const int u    = s >> 5;
        const int cofs = (s & 31) * 32;
        const int l    = c_tap[li][u] + 2 * t;
        const int p    = c_pl[li][u];
        const __bf16* xa = xh + ((size_t)l * BATCH + b0 + r2) * CIN + cofs;
        const __bf16* xb = xl + ((size_t)l * BATCH + b0 + r2) * CIN + cofs;
        const __bf16* wa = wh + ((size_t)p * COUT + o0 + r2) * CIN + cofs;
        const __bf16* wb = wl + ((size_t)p * COUT + o0 + r2) * CIN + cofs;
        ASYNC_COPY(&sAh[buf][r2 * 32 + k2 * 8],       xa + k2 * 8);
        ASYNC_COPY(&sAh[buf][r2 * 32 + (k2 + 1) * 8], xa + (k2 + 1) * 8);
        ASYNC_COPY(&sAl[buf][r2 * 32 + k2 * 8],       xb + k2 * 8);
        ASYNC_COPY(&sAl[buf][r2 * 32 + (k2 + 1) * 8], xb + (k2 + 1) * 8);
        ASYNC_COPY(&sBh[buf][r2 * 32 + k2 * 8],       wa + k2 * 8);
        ASYNC_COPY(&sBh[buf][r2 * 32 + (k2 + 1) * 8], wa + (k2 + 1) * 8);
        ASYNC_COPY(&sBl[buf][r2 * 32 + k2 * 8],       wb + k2 * 8);
        ASYNC_COPY(&sBl[buf][r2 * 32 + (k2 + 1) * 8], wb + (k2 + 1) * 8);
    };

    issue(0, 0);
    wait_async_0();
    __syncthreads();

    for (int s = 0; s < S; ++s) {
        const int buf = s & 1;
        if (s + 1 < S) issue(s + 1, buf ^ 1);

        // A fragments: lane holds row m=l16; k = hl*8+0..7 and 16+hl*8+0..7
        v16bf afh[2], afl[2];
#pragma unroll
        for (int i = 0; i < 2; ++i) {
            const int r = wm * 32 + i * 16 + l16;
            union { v16bf v; v8bf h[2]; } ta, tb;
            ta.h[0] = *(const v8bf*)&sAh[buf][r * 32 + hl * 8];
            ta.h[1] = *(const v8bf*)&sAh[buf][r * 32 + 16 + hl * 8];
            afh[i] = ta.v;
            tb.h[0] = *(const v8bf*)&sAl[buf][r * 32 + hl * 8];
            tb.h[1] = *(const v8bf*)&sAl[buf][r * 32 + 16 + hl * 8];
            afl[i] = tb.v;
        }
#pragma unroll
        for (int j = 0; j < 4; ++j) {
            // B fragment: lane holds col n=l16; k = hl*8+0..7 and 16+hl*8+0..7
            const int r = wn * 64 + j * 16 + l16;
            union { v16bf v; v8bf h[2]; } tbh, tbl;
            tbh.h[0] = *(const v8bf*)&sBh[buf][r * 32 + hl * 8];
            tbh.h[1] = *(const v8bf*)&sBh[buf][r * 32 + 16 + hl * 8];
            tbl.h[0] = *(const v8bf*)&sBl[buf][r * 32 + hl * 8];
            tbl.h[1] = *(const v8bf*)&sBl[buf][r * 32 + 16 + hl * 8];
#pragma unroll
            for (int i = 0; i < 2; ++i) {
                acc[i][j] = __builtin_amdgcn_wmma_f32_16x16x32_bf16(
                    false, afh[i], false, tbh.v, (short)0, acc[i][j], false, false);
                acc[i][j] = __builtin_amdgcn_wmma_f32_16x16x32_bf16(
                    false, afh[i], false, tbl.v, (short)0, acc[i][j], false, false);
                acc[i][j] = __builtin_amdgcn_wmma_f32_16x16x32_bf16(
                    false, afl[i], false, tbh.v, (short)0, acc[i][j], false, false);
            }
        }

        wait_async_0();        // prefetch for s+1 landed
        __syncthreads();       // everyone done reading buf / writes visible
    }

    // epilogue: D layout -> lane l: n = l%16, m = r + 8*(l/16) for VGPR r
    const int col = t * 6 + li;
#pragma unroll
    for (int i = 0; i < 2; ++i) {
#pragma unroll
        for (int j = 0; j < 4; ++j) {
            const int o_g = o0 + wn * 64 + j * 16 + l16;
#pragma unroll
            for (int r = 0; r < 8; ++r) {
                const int b_g = b0 + wm * 32 + i * 16 + r + 8 * hl;
                out[((size_t)b_g * COUT + o_g) * NCOL + col] = acc[i][j][r];
            }
        }
    }
}

// ---------------------------------------------------------------------------
extern "C" void kernel_launch(void* const* d_in, const int* in_sizes, int n_in,
                              void* d_out, int out_size, void* d_ws, size_t ws_size,
                              hipStream_t stream) {
    (void)in_sizes; (void)n_in; (void)ws_size;
    const float* x  = (const float*)d_in[0];
    const float* f2 = (const float*)d_in[1];
    const float* f3 = (const float*)d_in[2];
    float* out = (float*)d_out;

    char* ws = (char*)d_ws;
    const size_t XSZ = (size_t)LIN * BATCH * CIN * sizeof(__bf16);   // 9,437,184 B
    const size_t WSZ = (size_t)5 * COUT * CIN * sizeof(__bf16);      // 10,485,760 B
    __bf16* xh = (__bf16*)(ws);
    __bf16* xl = (__bf16*)(ws + XSZ);
    __bf16* wh = (__bf16*)(ws + 2 * XSZ);
    __bf16* wl = (__bf16*)(ws + 2 * XSZ + WSZ);

    const int nx = LIN * BATCH * CIN;
    split_x_kernel<<<(nx + 255) / 256, 256, 0, stream>>>(x, xh, xl, nx);
    const int nw = 5 * COUT * CIN;
    split_w_kernel<<<(nw + 255) / 256, 256, 0, stream>>>(f2, f3, wh, wl, nw);

    const int n4 = out_size / 4;
    zero_kernel<<<(n4 + 255) / 256, 256, 0, stream>>>((float4*)d_out, n4);

    // 42 m-tiles total across the 6 lengths (2*T each), 8 n-tiles of 128
    dim3 grid(42, COUT / 128);
    gemm_wmma_kernel<<<grid, 256, 0, stream>>>(xh, xl, wh, wl, out);
}
